// Indexer_66271345377673
// MI455X (gfx1250) — matrix operand order repaired and checked by reading, not compiled, not run on hardware
//
#include <hip/hip_runtime.h>
#include <hip/hip_bf16.h>

typedef __bf16 v16bf __attribute__((ext_vector_type(16)));
typedef __bf16 v8bf  __attribute__((ext_vector_type(8)));
typedef float  v8f   __attribute__((ext_vector_type(8)));

#define NEG_SCORE (-1e30f)

// ds_swizzle_b32 group-of-32 XOR exchange: offset = (xor<<10) | (or<<5) | and
template <int XMASK>
__device__ __forceinline__ float swz_xor(float x) {
    return __int_as_float(
        __builtin_amdgcn_ds_swizzle(__float_as_int(x), (XMASK << 10) | 0x1f));
}

// ---------------------------------------------------------------------------
// Kernel 1: Fast Walsh-Hadamard transform over last dim (D=128), bf16->bf16.
// One wave32 per row; lane holds elements {L, L+32, L+64, L+96}.
// Stages 1..16 via ds_swizzle XOR, stages 32/64 in-register. Scale = 128^-0.5.
// ---------------------------------------------------------------------------
template <int D>
__device__ __forceinline__ void fwht_stage(float& v0, float& v1, float& v2,
                                           float& v3, int lane) {
    float o0 = swz_xor<D>(v0), o1 = swz_xor<D>(v1);
    float o2 = swz_xor<D>(v2), o3 = swz_xor<D>(v3);
    bool hi = (lane & D) != 0;
    v0 = hi ? (o0 - v0) : (v0 + o0);
    v1 = hi ? (o1 - v1) : (v1 + o1);
    v2 = hi ? (o2 - v2) : (v2 + o2);
    v3 = hi ? (o3 - v3) : (v3 + o3);
}

__global__ __launch_bounds__(256, 1)
void fwht128_kernel(const __bf16* __restrict__ q,
                    const __bf16* __restrict__ k,
                    __bf16* __restrict__ qr,
                    __bf16* __restrict__ kr,
                    int qrows, int krows) {
    int wave = (int)((blockIdx.x * blockDim.x + threadIdx.x) >> 5);
    int lane = (int)(threadIdx.x & 31);
    if (wave >= qrows + krows) return;

    const __bf16* src;
    __bf16* dst;
    if (wave < qrows) { src = q + (size_t)wave * 128;            dst = qr + (size_t)wave * 128; }
    else              { src = k + (size_t)(wave - qrows) * 128;  dst = kr + (size_t)(wave - qrows) * 128; }

    float v0 = (float)src[lane];
    float v1 = (float)src[lane + 32];
    float v2 = (float)src[lane + 64];
    float v3 = (float)src[lane + 96];

    fwht_stage<1>(v0, v1, v2, v3, lane);
    fwht_stage<2>(v0, v1, v2, v3, lane);
    fwht_stage<4>(v0, v1, v2, v3, lane);
    fwht_stage<8>(v0, v1, v2, v3, lane);
    fwht_stage<16>(v0, v1, v2, v3, lane);
    // distance 32: (v0,v1), (v2,v3)
    float a = v0 + v1, b = v0 - v1, c = v2 + v3, d4 = v2 - v3;
    v0 = a; v1 = b; v2 = c; v3 = d4;
    // distance 64: (v0,v2), (v1,v3)
    a = v0 + v2; c = v0 - v2; b = v1 + v3; d4 = v1 - v3;
    v0 = a; v2 = c; v1 = b; v3 = d4;

    const float s = 0.08838834764831845f; // 128^-0.5
    dst[lane]      = (__bf16)(v0 * s);
    dst[lane + 32] = (__bf16)(v1 * s);
    dst[lane + 64] = (__bf16)(v2 * s);
    dst[lane + 96] = (__bf16)(v3 * s);
}

// ---------------------------------------------------------------------------
// Kernel 2: scores[t, s] = sum_h w[t,h] * relu( qr[t,h,:] . kr[s,:] ), masked.
// One block per query token t (4 waves). M-tile of 16 == the 16 heads of t.
// Each wave processes PAIRS of adjacent 16-wide key tiles: all 16 b128 loads
// issued up front, then 8 WMMAs in two independent D-chains drained with
// partial loadcnt waits. Head reduction: 8 acc VGPRs + ds_swizzle SWAPX16.
// __launch_bounds__(128,1): max VGPR budget so all B fragments stay live.
// ---------------------------------------------------------------------------
__device__ __forceinline__ void load_bfrag(const __bf16* __restrict__ brow,
                                           v16bf bfrag[4]) {
    #pragma unroll
    for (int c = 0; c < 4; ++c) {
        union { v16bf v; v8bf h[2]; } ub;
        ub.h[0] = *(const v8bf*)(brow + c * 32);
        ub.h[1] = *(const v8bf*)(brow + c * 32 + 8);
        bfrag[c] = ub.v;
    }
}

__device__ __forceinline__ float reduce_tile(const v8f& acc, const float wv[8]) {
    float p = 0.0f;
    #pragma unroll
    for (int r = 0; r < 8; ++r) p += wv[r] * fmaxf(acc[r], 0.0f);
    return p + swz_xor<16>(p);   // add other half-wave's partial (M 0-7 vs 8-15)
}

__global__ __launch_bounds__(128, 1)
void score_kernel(const __bf16* __restrict__ qr,
                  const __bf16* __restrict__ kr,
                  const float*  __restrict__ weights,
                  const int*    __restrict__ seq_lens,
                  int n_seq,
                  float* __restrict__ scores,
                  int T) {
    const int t      = (int)blockIdx.x;
    const int lane   = (int)(threadIdx.x & 31);
    const int wave   = (int)(threadIdx.x >> 5);
    const int nwaves = (int)(blockDim.x >> 5);

    // start of t's sequence in the packed KV (ragged causal window is [ks, t])
    int ks = 0;
    {
        int cum = 0;
        for (int b = 0; b < n_seq; ++b) {
            int len = seq_lens[b];
            if (t >= cum + len) cum += len; else break;
        }
        ks = cum;
    }

    const int half = (lane >> 4) & 1;   // which 16-lane half of the wave
    const int mrow = lane & 15;         // M row (head) for A, N column for B/C

    // Per-lane weights for the C rows this lane reduces (M = half*8 + r).
    float wv[8];
    #pragma unroll
    for (int r = 0; r < 8; ++r) wv[r] = weights[t * 16 + half * 8 + r];

    // Preload all 4 K-chunks of the A fragment (qr row block for token t).
    // A 16x32 layout (contiguous K, matching B): lanes 0-15 row M=lane hold K=0..15,
    // lanes 16-31 same rows hold K=16..31.
    const __bf16* arow = qr + ((size_t)t * 16 + mrow) * 128;
    v16bf afrag[4];
    #pragma unroll
    for (int c = 0; c < 4; ++c) {
        int base = c * 32 + half * 16;
        union { v16bf v; v8bf h[2]; } ua;
        ua.h[0] = *(const v8bf*)(arow + base);
        ua.h[1] = *(const v8bf*)(arow + base + 8);
        afrag[c] = ua.v;
    }

    const int ntiles = T >> 4;
    const int npairs = ntiles >> 1;

    // Main loop: 32-wide pairs of key tiles per wave
    for (int pt = wave; pt < npairs; pt += nwaves) {
        const int s0 = pt << 5;
        float totA = 0.0f, totB = 0.0f;
        const bool compute = (s0 <= t) && (s0 + 31 >= ks);   // wave-uniform
        if (compute) {
            // B layout (32x16 bf16): lanes 0-15 column N=lane, K=0..15
            // consecutive; lanes 16-31 same column, K=16..31.
            const __bf16* browA = kr + (size_t)(s0 + mrow) * 128 + half * 16;
            const __bf16* browB = browA + (size_t)16 * 128;

            // prefetch next pair's B rows for this wave
            int snext = s0 + (nwaves << 5);
            if (snext <= t && snext < T)
                __builtin_prefetch(kr + (size_t)(snext + mrow) * 128, 0, 1);

            v16bf ba[4], bb[4];
            load_bfrag(browA, ba);       // 16 x global_load_b128 up front
            load_bfrag(browB, bb);

            v8f accA = {}, accB = {};
            #pragma unroll
            for (int c = 0; c < 4; ++c) {
                accA = __builtin_amdgcn_wmma_f32_16x16x32_bf16(
                           false, afrag[c], false, ba[c], (short)0, accA,
                           false, false);
                accB = __builtin_amdgcn_wmma_f32_16x16x32_bf16(
                           false, afrag[c], false, bb[c], (short)0, accB,
                           false, false);
            }
            totA = reduce_tile(accA, wv);
            totB = reduce_tile(accB, wv);
        }
        if (lane < 16) {
            int sA = s0 + lane, sB = s0 + 16 + lane;
            bool vA = (sA >= ks) && (sA <= t);
            bool vB = (sB >= ks) && (sB <= t);
            scores[(size_t)t * T + sA] = vA ? totA : NEG_SCORE;
            scores[(size_t)t * T + sB] = vB ? totB : NEG_SCORE;
        }
    }

    // Tail: leftover single tile if ntiles is odd
    for (int st = (npairs << 1) + wave; st < ntiles; st += nwaves) {
        const int s0 = st << 4;
        float tot = 0.0f;
        const bool compute = (s0 <= t) && (s0 + 15 >= ks);
        if (compute) {
            const __bf16* brow = kr + (size_t)(s0 + mrow) * 128 + half * 16;
            v16bf bfrag[4];
            load_bfrag(brow, bfrag);
            v8f acc = {};
            #pragma unroll
            for (int c = 0; c < 4; ++c)
                acc = __builtin_amdgcn_wmma_f32_16x16x32_bf16(
                          false, afrag[c], false, bfrag[c], (short)0, acc,
                          false, false);
            tot = reduce_tile(acc, wv);
        }
        if (lane < 16) {
            int s = s0 + lane;
            bool valid = (s >= ks) && (s <= t);
            scores[(size_t)t * T + s] = valid ? tot : NEG_SCORE;
        }
    }
}

// ---------------------------------------------------------------------------
// Kernel 3: exact per-row top-K via full bitonic sort of 64-bit keys in LDS.
// key = sortable(value) << 32 | ~index  (descending value, ascending index
// tie-break -> matches jax.lax.top_k ordering).
// ---------------------------------------------------------------------------
__global__ __launch_bounds__(256, 1)
void topk_kernel(const float* __restrict__ scores,
                 float* __restrict__ out_vals,
                 int*   __restrict__ out_idx,
                 int T, int K, int N2) {
    extern __shared__ unsigned long long keys[];
    const int t   = (int)blockIdx.x;
    const int tid = (int)threadIdx.x;
    const int nth = (int)blockDim.x;
    const float* row = scores + (size_t)t * T;

    for (int i = tid; i < N2; i += nth) {
        if (i < T) {
            unsigned u = __float_as_uint(row[i]);
            u ^= (u >> 31) ? 0xFFFFFFFFu : 0x80000000u;      // sortable encoding
            keys[i] = ((unsigned long long)u << 32) | (unsigned)(~i);
        } else {
            keys[i] = 0ull;                                  // below any real key
        }
    }
    __syncthreads();

    for (int k = 2; k <= N2; k <<= 1) {
        for (int j = k >> 1; j > 0; j >>= 1) {
            for (int i = tid; i < N2; i += nth) {
                int ixj = i ^ j;
                if (ixj > i) {
                    unsigned long long a = keys[i], b = keys[ixj];
                    bool desc = ((i & k) == 0);
                    if ((a < b) == desc) { keys[i] = b; keys[ixj] = a; }
                }
            }
            __syncthreads();
        }
    }

    for (int i = tid; i < K; i += nth) {
        unsigned long long key = keys[i];
        unsigned su = (unsigned)(key >> 32);
        su ^= (su >> 31) ? 0x80000000u : 0xFFFFFFFFu;        // decode
        out_vals[(size_t)t * K + i] = __uint_as_float(su);
        out_idx [(size_t)t * K + i] = (int)(~(unsigned)(key & 0xFFFFFFFFu));
    }
}

// ---------------------------------------------------------------------------
extern "C" void kernel_launch(void* const* d_in, const int* in_sizes, int n_in,
                              void* d_out, int out_size, void* d_ws, size_t ws_size,
                              hipStream_t stream) {
    const __bf16* q       = (const __bf16*)d_in[0];   // [T, H, 128] bf16
    const __bf16* k       = (const __bf16*)d_in[1];   // [T, 128]    bf16
    const float*  weights = (const float*)d_in[2];    // [T, H]      f32
    const int*    seqlens = (const int*)d_in[3];      // [n_seq]     i32
    const int n_seq = in_sizes[3];

    const int D = 128;
    const int T = in_sizes[1] / D;
    const int H = in_sizes[0] / in_sizes[1];
    const int K = out_size / (2 * T);                 // top-k per row

    // workspace layout: qr | kr | scores
    char* ws = (char*)d_ws;
    size_t off_qr = 0;
    size_t off_kr = ((size_t)T * H * D * 2 + 255) & ~(size_t)255;
    size_t off_sc = ((off_kr + (size_t)T * D * 2) + 255) & ~(size_t)255;
    __bf16* qr     = (__bf16*)(ws + off_qr);
    __bf16* kr     = (__bf16*)(ws + off_kr);
    float*  scores = (float*)(ws + off_sc);

    float* out_vals = (float*)d_out;
    int*   out_idx  = (int*)((float*)d_out + (size_t)T * K);

    // 1) Hadamard rotation of q rows (T*H) and k rows (T): 8 waves / block
    {
        int rows = T * H + T;
        int blocks = (rows + 7) / 8;
        fwht128_kernel<<<blocks, 256, 0, stream>>>(q, k, qr, kr, T * H, T);
    }

    // 2) WMMA scores + causal ragged mask: one block (4 waves) per token
    score_kernel<<<T, 128, 0, stream>>>(qr, kr, weights, seqlens, n_seq, scores, T);

    // 3) per-row exact top-K (bitonic sort over padded power-of-two row)
    {
        int N2 = 1;
        while (N2 < T) N2 <<= 1;
        size_t shmem = (size_t)N2 * sizeof(unsigned long long);
        topk_kernel<<<T, 256, shmem, stream>>>(scores, out_vals, out_idx, T, K, N2);
    }
}
